// StackedLSTM_64793876628231
// MI455X (gfx1250) — compile-verified, matrix operand order checked by reading, not run on hardware
//
#include <hip/hip_runtime.h>

// Stacked LSTM (B=64, T=256, D=H=1024, L=2) for gfx1250 (CDNA5, wave32).
//
//   1. Convert weights/activations to bf16 once; fold bi+bh into one bias.
//   2. Per layer: one large parallel WMMA GEMM computes Xpre = X @ Wi^T + b
//      for ALL timesteps (off the recurrent critical path).
//   3. Per timestep: small WMMA GEMM h @ Wh^T. Each block owns a 32-wide
//      hidden slice for the FULL batch, so every Wh fragment is loaded from
//      global exactly once per step, staged via LDS. Each wave holds the 4
//      gate accumulators (I/F/G/O) for one 16x16 (batch x hidden) tile and
//      does the sigmoid/tanh update in registers.
//
// Matrix math: V_WMMA_F32_16X16X32_BF16 (fp32 accumulate).
// Pipelining: ldg(next K tile)->wmma(current)->ds_store(next)->barrier, with
// swapped LDS pointers and #pragma unroll 1 so the loop has ONE body and the
// accumulators stay register-tied through the WMMA (no phi copies).
// LDS tiles padded to row stride 40 bf16 (80 B) -> ds_load_b128 reads are
// bank-conflict-free (lane start banks 20*n mod 64 all distinct).

#define B_   64
#define T_   256
#define H_   1024
#define N4_  4096      // 4*H
#define L_   2
#define BT_  (B_ * T_) // 16384
#define LDP_ 40        // padded LDS row stride (elements) for 32-wide K tiles

typedef __attribute__((ext_vector_type(16))) __bf16 v16bf;
typedef __attribute__((ext_vector_type(8)))  __bf16 v8bf;
typedef __attribute__((ext_vector_type(8)))  float  v8f;

// ---------------------------------------------------------------------------
// Fragment layout (ISA 7.12.2, 16-bit A/B 16x32): lane l (<16) holds row l,
// K chunks [8h..8h+7] and [16+8h..16+8h+7] where h = lane>>4. Two contiguous
// 16B reads per lane.
// ---------------------------------------------------------------------------
__device__ __forceinline__ v16bf frag_lds(const __bf16* tile, int lane) {
  const int row  = lane & 15;
  const int half = lane >> 4;
  const __bf16* p = tile + row * LDP_ + half * 8;
  v8bf lo = *(const v8bf*)(p);        // K = 8h .. 8h+7
  v8bf hi = *(const v8bf*)(p + 16);   // K = 16+8h .. 16+8h+7
  return __builtin_shufflevector(lo, hi, 0, 1, 2, 3, 4, 5, 6, 7,
                                         8, 9, 10, 11, 12, 13, 14, 15);
}

__device__ __forceinline__ float sigm(float x) {
  return 1.0f / (1.0f + __expf(-x));
}

// ---------------------------------------------------------------------------
// Utility kernels
// ---------------------------------------------------------------------------
__global__ void f2bf_kernel(const float* __restrict__ src,
                            __bf16* __restrict__ dst, int n) {
  int i = blockIdx.x * blockDim.x + threadIdx.x;
  if (i < n) dst[i] = (__bf16)src[i];
}

__global__ void bias_sum_kernel(const float* __restrict__ bi,
                                const float* __restrict__ bh,
                                float* __restrict__ bs, int n) {
  int i = blockIdx.x * blockDim.x + threadIdx.x;
  if (i < n) bs[i] = bi[i] + bh[i];
}

__global__ void zero_f32_kernel(float* __restrict__ p, int n) {
  int i = blockIdx.x * blockDim.x + threadIdx.x;
  if (i < n) p[i] = 0.0f;
}

__global__ void zero_bf_kernel(__bf16* __restrict__ p, int n) {
  int i = blockIdx.x * blockDim.x + threadIdx.x;
  if (i < n) p[i] = (__bf16)0.0f;
}

__global__ void finalize_kernel(const float* __restrict__ hbuf,
                                const float* __restrict__ cbuf,
                                float* __restrict__ out) {
  int i = blockIdx.x * blockDim.x + threadIdx.x;
  const int LBH = L_ * B_ * H_;
  if (i < LBH) {
    out[(size_t)BT_ * H_ + i]       = hbuf[i];  // hT
    out[(size_t)BT_ * H_ + LBH + i] = cbuf[i];  // cT
  }
}

// ---------------------------------------------------------------------------
// Pre-GEMM: Xpre[BT,4H] = Xbf[BT,1024] @ W[4H,1024]^T + bsum   (fp32 out)
// grid = (N4/64, BT/128), block = 256 (8 waves).
// Block tile: 128(M) x 64(N); K staged 32 at a time through LDS, double
// buffered. Wave w owns M-rows [w*16..+15] and all 4 N-tiles (4 accums).
// ---------------------------------------------------------------------------
__global__ __launch_bounds__(256)
void pregemm_kernel(const __bf16* __restrict__ X, const __bf16* __restrict__ W,
                    const float* __restrict__ bsum, float* __restrict__ Xpre) {
  __shared__ __bf16 As0[128 * LDP_], As1[128 * LDP_];   // 10 KB each
  __shared__ __bf16 Bs0[64 * LDP_],  Bs1[64 * LDP_];    //  5 KB each
  const int tid   = threadIdx.x;
  const int lane  = tid & 31;
  const int wave  = tid >> 5;
  const int mbase = blockIdx.y * 128;
  const int nbase = blockIdx.x * 64;

  // Per-thread staging coordinates (3 x 16B chunks per thread per K tile)
  const int srow = tid >> 2;            // 0..63
  const int sq   = (tid & 3) * 8;
  const __bf16* gA0 = X + (size_t)(mbase + srow) * H_ + sq;       // A rows 0..63
  const __bf16* gA1 = X + (size_t)(mbase + srow + 64) * H_ + sq;  // A rows 64..127
  const __bf16* gB  = W + (size_t)(nbase + srow) * H_ + sq;       // B rows 0..63
  const int dA0 = srow * LDP_ + sq;
  const int dA1 = (srow + 64) * LDP_ + sq;
  const int dB  = srow * LDP_ + sq;

  v8f acc[4];
#pragma unroll
  for (int j = 0; j < 4; ++j) acc[j] = 0.0f;

  __bf16 *a_cur = As0, *a_nxt = As1, *b_cur = Bs0, *b_nxt = Bs1;

  // Prologue: stage K tile 0
  {
    v8bf r0 = *(const v8bf*)(gA0);
    v8bf r1 = *(const v8bf*)(gA1);
    v8bf r2 = *(const v8bf*)(gB);
    *(v8bf*)&a_cur[dA0] = r0;
    *(v8bf*)&a_cur[dA1] = r1;
    *(v8bf*)&b_cur[dB]  = r2;
  }
  __syncthreads();

#pragma unroll 1
  for (int kk = 0; kk < 31; ++kk) {
    const int kn = (kk + 1) * 32;
    // Issue next-tile global loads (in flight during the WMMAs below)
    v8bf r0 = *(const v8bf*)(gA0 + kn);
    v8bf r1 = *(const v8bf*)(gA1 + kn);
    v8bf r2 = *(const v8bf*)(gB + kn);

    v16bf a = frag_lds(a_cur + wave * 16 * LDP_, lane);
#pragma unroll
    for (int j = 0; j < 4; ++j) {
      v16bf b = frag_lds(b_cur + j * 16 * LDP_, lane);
      acc[j] = __builtin_amdgcn_wmma_f32_16x16x32_bf16(
          false, a, false, b, (short)0, acc[j], false, false);
    }

    *(v8bf*)&a_nxt[dA0] = r0;
    *(v8bf*)&a_nxt[dA1] = r1;
    *(v8bf*)&b_nxt[dB]  = r2;
    __syncthreads();

    __bf16* t;
    t = a_cur; a_cur = a_nxt; a_nxt = t;
    t = b_cur; b_cur = b_nxt; b_nxt = t;
  }
  // Epilogue K tile (kk = 31)
  {
    v16bf a = frag_lds(a_cur + wave * 16 * LDP_, lane);
#pragma unroll
    for (int j = 0; j < 4; ++j) {
      v16bf b = frag_lds(b_cur + j * 16 * LDP_, lane);
      acc[j] = __builtin_amdgcn_wmma_f32_16x16x32_bf16(
          false, a, false, b, (short)0, acc[j], false, false);
    }
  }

  const int half = lane >> 4;
  const int ncol = lane & 15;
  const int mw   = mbase + wave * 16;
#pragma unroll
  for (int j = 0; j < 4; ++j) {
    const int n = nbase + j * 16 + ncol;
    const float bv = bsum[n];
#pragma unroll
    for (int r = 0; r < 8; ++r) {
      const int m = mw + r + 8 * half;        // C/D layout: VGPR r, lane half
      Xpre[(size_t)m * N4_ + n] = acc[j][r] + bv;
    }
  }
}

// ---------------------------------------------------------------------------
// Recurrent step: gates = Xpre[:,t,:] + h @ Wh^T ; then cell update.
// grid = 32 blocks x 256 threads (8 waves). Block b owns hidden cols
// [b*32, b*32+32) for the FULL batch: waves -> (mtile = w>>1, jj = w&1).
// B rows staged: 8 groups of 16 = {gate g, jtile jj} -> Wh read once/step.
// A rows staged: all 64 h rows for the current 32-wide K slab.
// ---------------------------------------------------------------------------
__global__ __launch_bounds__(256)
void lstm_step_kernel(const __bf16* __restrict__ hin,
                      const __bf16* __restrict__ Wh,
                      const float* __restrict__ Xpre,
                      float* __restrict__ cbuf,
                      __bf16* __restrict__ hout,
                      __bf16* __restrict__ seq_bf,   // layer-0 output (bf16) or null
                      float* __restrict__ seq_f32,   // top-layer output or null
                      float* __restrict__ hbuf,      // fp32 h (for hT)
                      int t) {
  __shared__ __bf16 As0[64 * LDP_],  As1[64 * LDP_];    //  5 KB each
  __shared__ __bf16 Bs0[128 * LDP_], Bs1[128 * LDP_];   // 10 KB each
  const int tid   = threadIdx.x;
  const int lane  = tid & 31;
  const int wave  = tid >> 5;            // 0..7
  const int mtile = wave >> 1;           // 0..3  (batch/16)
  const int jj    = wave & 1;            // 0..1
  const int jbase = blockIdx.x * 2;      // jtile pair

  // Staging coords: A 1 chunk + B 2 chunks per thread per K tile.
  const int srow = tid >> 2;             // 0..63
  const int sq   = (tid & 3) * 8;
  const __bf16* gA = hin + (size_t)srow * H_ + sq;
  const int dA = srow * LDP_ + sq;
  // B local rows: srow (0..63) and srow+64 (64..127); local row -> Wh row.
  auto growf = [&](int row) {
    const int grp = row >> 4;            // 0..7 = {g, j}
    return (grp >> 1) * H_ + (jbase + (grp & 1)) * 16 + (row & 15);
  };
  const __bf16* gB0 = Wh + (size_t)growf(srow) * H_ + sq;
  const __bf16* gB1 = Wh + (size_t)growf(srow + 64) * H_ + sq;
  const int dB0 = srow * LDP_ + sq;
  const int dB1 = (srow + 64) * LDP_ + sq;

  v8f acc[4];
#pragma unroll
  for (int g = 0; g < 4; ++g) acc[g] = 0.0f;

  __bf16 *a_cur = As0, *a_nxt = As1, *b_cur = Bs0, *b_nxt = Bs1;

  {
    v8bf r0 = *(const v8bf*)(gA);
    v8bf r1 = *(const v8bf*)(gB0);
    v8bf r2 = *(const v8bf*)(gB1);
    *(v8bf*)&a_cur[dA]  = r0;
    *(v8bf*)&b_cur[dB0] = r1;
    *(v8bf*)&b_cur[dB1] = r2;
  }
  __syncthreads();

#pragma unroll 1
  for (int kk = 0; kk < 31; ++kk) {
    const int kn = (kk + 1) * 32;
    v8bf r0 = *(const v8bf*)(gA + kn);
    v8bf r1 = *(const v8bf*)(gB0 + kn);
    v8bf r2 = *(const v8bf*)(gB1 + kn);

    v16bf a = frag_lds(a_cur + mtile * 16 * LDP_, lane);
#pragma unroll
    for (int g = 0; g < 4; ++g) {
      v16bf b = frag_lds(b_cur + (g * 2 + jj) * 16 * LDP_, lane);
      acc[g] = __builtin_amdgcn_wmma_f32_16x16x32_bf16(
          false, a, false, b, (short)0, acc[g], false, false);
    }

    *(v8bf*)&a_nxt[dA]  = r0;
    *(v8bf*)&b_nxt[dB0] = r1;
    *(v8bf*)&b_nxt[dB1] = r2;
    __syncthreads();

    __bf16* tp;
    tp = a_cur; a_cur = a_nxt; a_nxt = tp;
    tp = b_cur; b_cur = b_nxt; b_nxt = tp;
  }
  {
    v16bf a = frag_lds(a_cur + mtile * 16 * LDP_, lane);
#pragma unroll
    for (int g = 0; g < 4; ++g) {
      v16bf b = frag_lds(b_cur + (g * 2 + jj) * 16 * LDP_, lane);
      acc[g] = __builtin_amdgcn_wmma_f32_16x16x32_bf16(
          false, a, false, b, (short)0, acc[g], false, false);
    }
  }

  const int half = lane >> 4;
  const int n    = (jbase + jj) * 16 + (lane & 15);
#pragma unroll
  for (int r = 0; r < 8; ++r) {
    const int m = mtile * 16 + r + 8 * half;               // batch index
    const size_t xrow = (size_t)(m * T_ + t) * N4_;
    const float pI = acc[0][r] + Xpre[xrow + 0 * H_ + n];
    const float pF = acc[1][r] + Xpre[xrow + 1 * H_ + n];
    const float pG = acc[2][r] + Xpre[xrow + 2 * H_ + n];
    const float pO = acc[3][r] + Xpre[xrow + 3 * H_ + n];

    const size_t ci = (size_t)m * H_ + n;
    const float c_new = sigm(pF) * cbuf[ci] + sigm(pI) * tanhf(pG);
    const float h_new = sigm(pO) * tanhf(c_new);

    cbuf[ci] = c_new;
    hout[ci] = (__bf16)h_new;
    hbuf[ci] = h_new;
    if (seq_bf)  seq_bf[(size_t)(m * T_ + t) * H_ + n] = (__bf16)h_new;
    if (seq_f32) seq_f32[(size_t)(m * T_ + t) * H_ + n] = h_new;
  }
}

// ---------------------------------------------------------------------------
// Host orchestration
// ---------------------------------------------------------------------------
extern "C" void kernel_launch(void* const* d_in, const int* in_sizes, int n_in,
                              void* d_out, int out_size, void* d_ws, size_t ws_size,
                              hipStream_t stream) {
  const float* Xt = (const float*)d_in[0];   // [B,T,D]
  const float* Wi = (const float*)d_in[1];   // [L,4H,D]
  const float* Wh = (const float*)d_in[2];   // [L,4H,H]
  const float* bi = (const float*)d_in[3];   // [L,4H]
  const float* bh = (const float*)d_in[4];   // [L,4H]
  float* out = (float*)d_out;                // [B,T,H] ++ hT[L,B,H] ++ cT[L,B,H]

  // Workspace carve (needs ~370 MB)
  char* p = (char*)d_ws;
  auto carve = [&](size_t bytes) -> void* {
    void* r = (void*)p;
    p += (bytes + 255) & ~(size_t)255;
    return r;
  };
  const size_t WELEM = (size_t)L_ * N4_ * H_;     // 8.4M per weight tensor
  __bf16* Wibf = (__bf16*)carve(WELEM * 2);
  __bf16* Whbf = (__bf16*)carve(WELEM * 2);
  float*  bsum = (float*) carve((size_t)L_ * N4_ * 4);
  __bf16* Xbf1 = (__bf16*)carve((size_t)BT_ * H_ * 2);
  __bf16* Xbf2 = (__bf16*)carve((size_t)BT_ * H_ * 2);
  float*  Xpre = (float*) carve((size_t)BT_ * N4_ * 4);   // 268 MB
  __bf16* hb0  = (__bf16*)carve((size_t)B_ * H_ * 2);
  __bf16* hb1  = (__bf16*)carve((size_t)B_ * H_ * 2);
  float*  cbuf = (float*) carve((size_t)L_ * B_ * H_ * 4);
  float*  hbuf = (float*) carve((size_t)L_ * B_ * H_ * 4);

  // 1. Conversions
  {
    int n = (int)WELEM;
    f2bf_kernel<<<(n + 255) / 256, 256, 0, stream>>>(Wi, Wibf, n);
    f2bf_kernel<<<(n + 255) / 256, 256, 0, stream>>>(Wh, Whbf, n);
    int nb = L_ * N4_;
    bias_sum_kernel<<<(nb + 255) / 256, 256, 0, stream>>>(bi, bh, bsum, nb);
    int nx = BT_ * H_;
    f2bf_kernel<<<(nx + 255) / 256, 256, 0, stream>>>(Xt, Xbf1, nx);
    int nc = L_ * B_ * H_;
    zero_f32_kernel<<<(nc + 255) / 256, 256, 0, stream>>>(cbuf, nc);
  }

  // 2. Layers
  for (int l = 0; l < L_; ++l) {
    const __bf16* Xin = (l == 0) ? Xbf1 : Xbf2;
    const __bf16* Wil = Wibf + (size_t)l * N4_ * H_;
    const __bf16* Whl = Whbf + (size_t)l * N4_ * H_;
    const float*  bsl = bsum + (size_t)l * N4_;
    float* cl = cbuf + (size_t)l * B_ * H_;
    float* hl = hbuf + (size_t)l * B_ * H_;

    {
      int nh = B_ * H_;
      zero_bf_kernel<<<(nh + 255) / 256, 256, 0, stream>>>(hb0, nh);
    }

    // Parallel input projection for all timesteps
    pregemm_kernel<<<dim3(N4_ / 64, BT_ / 128), 256, 0, stream>>>(
        Xin, Wil, bsl, Xpre);

    // Sequential recurrence, ping-pong h
    for (int t = 0; t < T_; ++t) {
      const __bf16* hin = (t & 1) ? hb1 : hb0;
      __bf16*       ho  = (t & 1) ? hb0 : hb1;
      lstm_step_kernel<<<32, 256, 0, stream>>>(
          hin, Whl, Xpre, cl, ho,
          (l == 0) ? Xbf2 : (__bf16*)nullptr,
          (l == L_ - 1) ? out : (float*)nullptr,
          hl, t);
    }
  }

  // 3. hT / cT tail
  {
    int n = L_ * B_ * H_;
    finalize_kernel<<<(n + 255) / 256, 256, 0, stream>>>(hbuf, cbuf, out);
  }
}
